// DynamicReductionNetwork_82197084111199
// MI455X (gfx1250) — matrix operand order repaired and checked by reading, not compiled, not run on hardware
//
#include <hip/hip_runtime.h>
#include <cstddef>
#include <cstdint>

#define NB   16
#define NN   1024
#define KNN  16
#define HID  64
#define NKE  (NN*KNN)      // 16384 directed kNN edges per graph
#define EDG  (2*NKE)       // 32768 edges after to_undirected concat
#define EPSC 1e-5f
#define NEGF (-1e30f)
#define INF_F __builtin_inff()

typedef __attribute__((ext_vector_type(16))) __bf16  v16bf;
typedef __attribute__((ext_vector_type(8)))  float   v8f;

static __device__ __forceinline__ v8f wmma_bf16(v16bf a, v16bf b, v8f c) {
  // D = A(16x32 bf16) x B(32x16 bf16) + C(16x16 f32)
  return __builtin_amdgcn_wmma_f32_16x16x32_bf16(false, a, false, b, (short)0, c, false, false);
}

static __device__ __forceinline__ float eluf(float x) { return x > 0.f ? x : (expf(x) - 1.f); }

// order-preserving float -> unsigned key (for atomicMax-based segment max)
static __device__ __forceinline__ unsigned fkey(float f) {
  unsigned u = __float_as_uint(f);
  return (u & 0x80000000u) ? ~u : (u | 0x80000000u);
}
static __device__ __forceinline__ float unfkey(unsigned k) {
  unsigned u = (k & 0x80000000u) ? (k & 0x7fffffffu) : ~k;
  return __uint_as_float(u);
}

// ---------------------------------------------------------------------------
// Stage 1: input MLP 5 -> 32 -> 64 -> 64 (ELU), weights staged in LDS.
// Also emits bf16 mirror and squared norms for the WMMA Gram-matrix stage.
// ---------------------------------------------------------------------------
__global__ void k_input_mlp(const float* __restrict__ x, const float* __restrict__ dn,
                            const float* __restrict__ w1, const float* __restrict__ b1,
                            const float* __restrict__ w2, const float* __restrict__ b2,
                            const float* __restrict__ w3, const float* __restrict__ b3,
                            float* __restrict__ h, __bf16* __restrict__ hb,
                            float* __restrict__ d2, int* __restrict__ valid) {
  __shared__ float sm[6480];
  const int tid = threadIdx.x;
  for (int i = tid; i < 5;    i += 128) sm[i]       = dn[i];
  for (int i = tid; i < 160;  i += 128) sm[8+i]     = w1[i];
  for (int i = tid; i < 32;   i += 128) sm[168+i]   = b1[i];
  for (int i = tid; i < 2048; i += 128) sm[200+i]   = w2[i];
  for (int i = tid; i < 64;   i += 128) sm[2248+i]  = b2[i];
  for (int i = tid; i < 4096; i += 128) sm[2312+i]  = w3[i];
  for (int i = tid; i < 64;   i += 128) sm[6408+i]  = b3[i];
  __syncthreads();
  const int node = blockIdx.x * 128 + tid;   // < NB*NN
  float xm[5];
  for (int j = 0; j < 5; ++j) xm[j] = x[node*5+j] * sm[j];
  float t1[32];
  for (int o = 0; o < 32; ++o) {
    float s = sm[168+o];
    for (int j = 0; j < 5; ++j) s += xm[j] * sm[8+o*5+j];
    t1[o] = eluf(s);
  }
  float t2[64];
  for (int o = 0; o < 64; ++o) {
    float s = sm[2248+o];
    for (int j = 0; j < 32; ++j) s += t1[j] * sm[200+o*32+j];
    t2[o] = eluf(s);
  }
  float s2 = 0.f;
  for (int o = 0; o < 64; ++o) {
    float s = sm[6408+o];
    for (int j = 0; j < 64; ++j) s += t2[j] * sm[2312+o*64+j];
    s = eluf(s);
    h[node*HID+o] = s;
    hb[node*HID+o] = (__bf16)s;
    s2 += s * s;
  }
  d2[node] = s2;
  valid[node] = 1;
}

// ---------------------------------------------------------------------------
// Stage 2: kNN. Workgroup = (graph, 16-row block). 4 waves build the 16x1024
// distance slab via bf16 WMMA Gram tiles (K=64 = 2 chained 16x16x32 WMMAs),
// then 16 threads do serial top-16 selection per row.
// Operand k order used here (contiguous, not AMD's table order): A lane half holds 16 consecutive K for row m = lane&15;
// B lane half holds 16 consecutive K for column n = lane&15.
// ---------------------------------------------------------------------------
__global__ void k_knn(const __bf16* __restrict__ hb, const float* __restrict__ d2,
                      const int* __restrict__ valid, int* __restrict__ knn) {
  __shared__ float drow[16][NN];            // 64 KB
  const int b = blockIdx.y, r0 = blockIdx.x * 16;
  const int tid = threadIdx.x, lane = tid & 31, wv = tid >> 5;
  const int m = lane & 15, hi = lane >> 4;
  const __bf16* Hb = hb + (size_t)b * NN * HID;
  const float* d2g = d2 + b * NN;
  const int* vg = valid + b * NN;
  v16bf a0, a1;
  {
    const __bf16* ar = Hb + (size_t)(r0 + m) * HID;
    for (int j = 0; j < 16; ++j) {
      int k = hi*16 + j;
      a0[j] = ar[k];
      a1[j] = ar[32 + k];
    }
  }
  for (int ct = wv; ct < 64; ct += 4) {
    const int c0 = ct * 16;
    const __bf16* br = Hb + (size_t)(c0 + m) * HID;
    v16bf b0, b1;
    for (int j = 0; j < 16; ++j) {
      b0[j] = br[hi*16 + j];
      b1[j] = br[32 + hi*16 + j];
    }
    v8f acc = {};
    acc = wmma_bf16(a0, b0, acc);
    acc = wmma_bf16(a1, b1, acc);
    for (int r = 0; r < 8; ++r) {
      int row = r + hi*8, grow = r0 + row, gcol = c0 + m;
      float dv = d2g[grow] + d2g[gcol] - 2.f * acc[r];
      bool bad = (grow == gcol) | (!vg[grow]) | (!vg[gcol]);
      drow[row][gcol] = bad ? INF_F : dv;
    }
  }
  __syncthreads();
  if (tid < 16) {
    const int row = tid;
    int* ko = knn + ((size_t)(b * NN + r0 + row)) * KNN;
    for (int k = 0; k < KNN; ++k) {
      float bd = drow[row][0]; int bi = 0;
      for (int j = 1; j < NN; ++j) {
        float v = drow[row][j];
        if (v < bd) { bd = v; bi = j; }
      }
      ko[k] = bi;
      drow[row][bi] = INF_F;
    }
  }
}

// ---------------------------------------------------------------------------
// Stage 3: to_undirected edge build with mutual-kNN dedup mask.
// ---------------------------------------------------------------------------
__global__ void k_build_edges(const int* __restrict__ knn, const int* __restrict__ valid,
                              int* __restrict__ src, int* __restrict__ dst, int* __restrict__ em) {
  const int b = blockIdx.y;
  const int t = blockIdx.x * 256 + threadIdx.x;   // < NKE
  const int* kb = knn + (size_t)b * NN * KNN;
  const int* vg = valid + b * NN;
  const int i = t >> 4, k = t & 15;
  const int s = kb[i * KNN + k];
  const int v1 = vg[i] && vg[s];
  int mut = 0;
  for (int kk = 0; kk < KNN; ++kk) mut |= (kb[s * KNN + kk] == i);
  int* sb = src + (size_t)b * EDG;
  int* db = dst + (size_t)b * EDG;
  int* eb = em  + (size_t)b * EDG;
  sb[t] = s;        db[t] = i;        eb[t] = v1;
  sb[NKE + t] = i;  db[NKE + t] = s;  eb[NKE + t] = v1 && !mut;
}

__global__ void k_zero(float* __restrict__ p, int n) {
  int i = blockIdx.x * 256 + threadIdx.x;
  if (i < n) p[i] = 0.f;
}

// ---------------------------------------------------------------------------
// Stage 4: EdgeConv. 64 edges per block. Messages [xi, xj-xi] staged as bf16
// in LDS, two WMMA GEMMs (128->96, 96->64) with BN+ELU epilogues, masked
// atomic segment-sum into the per-node accumulator.
// ---------------------------------------------------------------------------
__global__ void k_edgeconv(const float* __restrict__ h,
                           const int* __restrict__ src, const int* __restrict__ dst,
                           const int* __restrict__ em,
                           const float* __restrict__ w1, const float* __restrict__ b1,
                           const float* __restrict__ g1, const float* __restrict__ be1,
                           const float* __restrict__ w2, const float* __restrict__ b2,
                           const float* __restrict__ g2, const float* __restrict__ be2,
                           float* __restrict__ out) {
  __shared__ __bf16 msg[64][128];   // 16 KB
  __shared__ __bf16 mid[64][96];    // 12 KB
  const int b = blockIdx.y, e0 = blockIdx.x * 64;
  const int tid = threadIdx.x, lane = tid & 31, wv = tid >> 5;
  const int m = lane & 15, hi = lane >> 4;
  const float bnc = rsqrtf(1.f + EPSC);
  const int* sb = src + (size_t)b * EDG + e0;
  const int* db = dst + (size_t)b * EDG + e0;
  const int* eb = em  + (size_t)b * EDG + e0;
  const float* Hh = h + (size_t)b * NN * HID;

  for (int idx = tid; idx < 64 * 128; idx += 128) {
    int e = idx >> 7, f = idx & 127;
    int de = db[e], se = sb[e];
    float v = (f < 64) ? Hh[de*HID + f]
                       : (Hh[se*HID + (f - 64)] - Hh[de*HID + (f - 64)]);
    msg[e][f] = (__bf16)v;
  }
  __syncthreads();

  // GEMM1: [64 x 128] x [128 x 96] ; 4 m-tiles x 6 n-tiles, 4 k-chunks
  for (int t = wv; t < 24; t += 4) {
    int mt = t / 6, nt = t % 6;
    v8f acc = {};
    for (int kc = 0; kc < 4; ++kc) {
      v16bf a, bm;
      for (int j = 0; j < 16; ++j) {
        int ka = kc*32 + hi*16 + j;
        a[j]  = msg[mt*16 + m][ka];
        bm[j] = (__bf16)w1[(nt*16 + m) * 128 + kc*32 + hi*16 + j];
      }
      acc = wmma_bf16(a, bm, acc);
    }
    const int nc = nt*16 + m;
    const float sc = g1[nc] * bnc, bo = be1[nc], bia = b1[nc];
    for (int r = 0; r < 8; ++r) {
      int row = r + hi*8;
      float v = eluf((acc[r] + bia) * sc + bo);
      mid[mt*16 + row][nc] = (__bf16)v;
    }
  }
  __syncthreads();

  // GEMM2: [64 x 96] x [96 x 64] ; 4 m-tiles x 4 n-tiles, 3 k-chunks
  for (int t = wv; t < 16; t += 4) {
    int mt = t >> 2, nt = t & 3;
    v8f acc = {};
    for (int kc = 0; kc < 3; ++kc) {
      v16bf a, bm;
      for (int j = 0; j < 16; ++j) {
        int ka = kc*32 + hi*16 + j;
        a[j]  = mid[mt*16 + m][ka];
        bm[j] = (__bf16)w2[(nt*16 + m) * 96 + kc*32 + hi*16 + j];
      }
      acc = wmma_bf16(a, bm, acc);
    }
    const int nc = nt*16 + m;
    const float sc = g2[nc] * bnc, bo = be2[nc], bia = b2[nc];
    for (int r = 0; r < 8; ++r) {
      int row = r + hi*8;
      int e = mt*16 + row;
      float v = eluf((acc[r] + bia) * sc + bo);
      if (eb[e]) atomicAdd(&out[((size_t)b * NN + db[e]) * HID + nc], v);
    }
  }
}

// ---------------------------------------------------------------------------
// Stage 5: graclus-style pooling, one 1024-thread block per graph.
// deg -> edge weights -> segment max (monotone-key atomicMax) -> best/src
// segment-min -> mutual matching -> per-cluster feature max -> re-emit
// h / bf16 mirror / d2 / valid for round 2.
// ---------------------------------------------------------------------------
__global__ void k_pool(const int* __restrict__ src, const int* __restrict__ dst,
                       const int* __restrict__ em,
                       float* __restrict__ h, int* __restrict__ valid,
                       float* __restrict__ wbuf, unsigned* __restrict__ pxkey,
                       __bf16* __restrict__ hb, float* __restrict__ d2) {
  __shared__ float    deg[NN];
  __shared__ unsigned wkey[NN];
  __shared__ int      best[NN];
  __shared__ int      root[NN];
  const int b = blockIdx.x, tid = threadIdx.x;
  const int* sb = src + (size_t)b * EDG;
  const int* db = dst + (size_t)b * EDG;
  const int* eb = em  + (size_t)b * EDG;
  float* Hh = h + (size_t)b * NN * HID;
  int* vg = valid + b * NN;
  float* wb = wbuf + (size_t)b * EDG;
  unsigned* px = pxkey + (size_t)b * NN * HID;
  __bf16* hbp = hb + (size_t)b * NN * HID;
  float* d2b = d2 + b * NN;

  for (int i = tid; i < NN; i += 1024) { deg[i] = 0.f; wkey[i] = fkey(NEGF); best[i] = NN; }
  for (int i = tid; i < NN * HID; i += 1024) px[i] = fkey(NEGF);
  __syncthreads();

  for (int e = tid; e < EDG; e += 1024)
    if (eb[e]) atomicAdd(&deg[db[e]], 1.f);
  __syncthreads();

  for (int e = tid; e < EDG; e += 1024) {
    float w = NEGF;
    if (eb[e]) {
      int s = sb[e], d = db[e];
      float acc = 1e-12f;
      for (int f = 0; f < HID; ++f) { float dd = Hh[s*HID+f] - Hh[d*HID+f]; acc += dd * dd; }
      float attr = sqrtf(acc);
      w = attr * (1.f / fmaxf(deg[d], 1.f) + 1.f / fmaxf(deg[s], 1.f));
      atomicMax(&wkey[d], fkey(w));
    }
    wb[e] = w;
  }
  __syncthreads();

  for (int e = tid; e < EDG; e += 1024)
    if (eb[e] && wb[e] >= unfkey(wkey[db[e]])) atomicMin(&best[db[e]], sb[e]);
  __syncthreads();

  for (int i = tid; i < NN; i += 1024) {
    int bi = best[i];
    bool has = bi < NN;
    int pb = best[has ? bi : 0];
    bool matched = has && (pb == i);
    root[i] = matched ? (i < bi ? i : bi) : i;
  }
  __syncthreads();

  for (int i = tid; i < NN; i += 1024) {
    if (vg[i]) {
      int r = root[i];
      for (int f = 0; f < HID; ++f) atomicMax(&px[r*HID + f], fkey(Hh[i*HID + f]));
    }
  }
  __threadfence_block();
  __syncthreads();

  for (int i = tid; i < NN; i += 1024) {
    bool nv = vg[i] && (root[i] == i);
    float s2 = 0.f;
    for (int f = 0; f < HID; ++f) {
      float v = nv ? unfkey(px[i*HID + f]) : 0.f;
      Hh[i*HID + f] = v;
      hbp[i*HID + f] = (__bf16)v;
      s2 += v * v;
    }
    d2b[i] = s2;
    vg[i] = nv ? 1 : 0;
  }
}

// ---------------------------------------------------------------------------
// Stage 6: global max over valid nodes, then tiny output MLP.
// ---------------------------------------------------------------------------
__global__ void k_gmax(const float* __restrict__ h, const int* __restrict__ valid,
                       float* __restrict__ g) {
  const int b = blockIdx.x, f = threadIdx.x;
  const float* Hh = h + (size_t)b * NN * HID;
  const int* vg = valid + b * NN;
  float mx = NEGF;
  for (int i = 0; i < NN; ++i)
    if (vg[i]) mx = fmaxf(mx, Hh[i*HID + f]);
  g[b*HID + f] = mx;
}

__global__ void k_outmlp(const float* __restrict__ g,
                         const float* __restrict__ w1, const float* __restrict__ b1,
                         const float* __restrict__ g1, const float* __restrict__ be1,
                         const float* __restrict__ w2, const float* __restrict__ b2,
                         const float* __restrict__ w3, const float* __restrict__ b3,
                         float* __restrict__ out) {
  const int b = threadIdx.x;
  if (b >= NB) return;
  const float bnc = rsqrtf(1.f + EPSC);
  float t1[64], t2[32];
  for (int o = 0; o < 64; ++o) {
    float s = b1[o];
    for (int j = 0; j < 64; ++j) s += g[b*64 + j] * w1[o*64 + j];
    t1[o] = eluf(s * (g1[o] * bnc) + be1[o]);
  }
  for (int o = 0; o < 32; ++o) {
    float s = b2[o];
    for (int j = 0; j < 64; ++j) s += t1[j] * w2[o*64 + j];
    t2[o] = eluf(s);
  }
  float s = b3[0];
  for (int j = 0; j < 32; ++j) s += t2[j] * w3[j];
  out[b] = s;
}

// ---------------------------------------------------------------------------
extern "C" void kernel_launch(void* const* d_in, const int* in_sizes, int n_in,
                              void* d_out, int out_size, void* d_ws, size_t ws_size,
                              hipStream_t stream) {
  (void)in_sizes; (void)n_in; (void)out_size; (void)ws_size;
  const float* x     = (const float*)d_in[0];
  const float* dn    = (const float*)d_in[1];
  const float* i_w1  = (const float*)d_in[2];  const float* i_b1  = (const float*)d_in[3];
  const float* i_w2  = (const float*)d_in[4];  const float* i_b2  = (const float*)d_in[5];
  const float* i_w3  = (const float*)d_in[6];  const float* i_b3  = (const float*)d_in[7];
  const float* c1_w1 = (const float*)d_in[8];  const float* c1_b1 = (const float*)d_in[9];
  const float* c1_g1 = (const float*)d_in[10]; const float* c1_be1= (const float*)d_in[11];
  const float* c1_w2 = (const float*)d_in[12]; const float* c1_b2 = (const float*)d_in[13];
  const float* c1_g2 = (const float*)d_in[14]; const float* c1_be2= (const float*)d_in[15];
  const float* c2_w1 = (const float*)d_in[16]; const float* c2_b1 = (const float*)d_in[17];
  const float* c2_g1 = (const float*)d_in[18]; const float* c2_be1= (const float*)d_in[19];
  const float* c2_w2 = (const float*)d_in[20]; const float* c2_b2 = (const float*)d_in[21];
  const float* c2_g2 = (const float*)d_in[22]; const float* c2_be2= (const float*)d_in[23];
  const float* o_w1  = (const float*)d_in[24]; const float* o_b1  = (const float*)d_in[25];
  const float* o_g1  = (const float*)d_in[26]; const float* o_be1 = (const float*)d_in[27];
  const float* o_w2  = (const float*)d_in[28]; const float* o_b2  = (const float*)d_in[29];
  const float* o_w3  = (const float*)d_in[30]; const float* o_b3  = (const float*)d_in[31];
  float* outp = (float*)d_out;

  char* w = (char*)d_ws;
  auto take = [&](size_t bytes) -> char* {
    char* p = w;
    w += (bytes + 255) & ~size_t(255);
    return p;
  };
  float*    hA    = (float*)   take((size_t)NB*NN*HID*4);   // node features (ping)
  float*    hC    = (float*)   take((size_t)NB*NN*HID*4);   // node features (pong / conv accum)
  __bf16*   hbm   = (__bf16*)  take((size_t)NB*NN*HID*2);   // bf16 mirror for WMMA
  float*    d2    = (float*)   take((size_t)NB*NN*4);       // squared norms
  int*      valid = (int*)     take((size_t)NB*NN*4);
  int*      knn   = (int*)     take((size_t)NB*NN*KNN*4);
  int*      srcb  = (int*)     take((size_t)NB*EDG*4);
  int*      dstb  = (int*)     take((size_t)NB*EDG*4);
  int*      emb   = (int*)     take((size_t)NB*EDG*4);
  float*    wbuf  = (float*)   take((size_t)NB*EDG*4);
  unsigned* pxk   = (unsigned*)take((size_t)NB*NN*HID*4);
  float*    gfeat = (float*)   take((size_t)NB*HID*4);

  k_input_mlp<<<dim3((NB*NN)/128), dim3(128), 0, stream>>>(
      x, dn, i_w1, i_b1, i_w2, i_b2, i_w3, i_b3, hA, hbm, d2, valid);

  k_knn<<<dim3(NN/16, NB), dim3(128), 0, stream>>>(hbm, d2, valid, knn);
  k_build_edges<<<dim3(NKE/256, NB), dim3(256), 0, stream>>>(knn, valid, srcb, dstb, emb);

  k_zero<<<dim3((NB*NN*HID)/256), dim3(256), 0, stream>>>(hC, NB*NN*HID);
  k_edgeconv<<<dim3(EDG/64, NB), dim3(128), 0, stream>>>(
      hA, srcb, dstb, emb,
      c1_w1, c1_b1, c1_g1, c1_be1, c1_w2, c1_b2, c1_g2, c1_be2, hC);

  k_pool<<<dim3(NB), dim3(1024), 0, stream>>>(srcb, dstb, emb, hC, valid, wbuf, pxk, hbm, d2);

  k_knn<<<dim3(NN/16, NB), dim3(128), 0, stream>>>(hbm, d2, valid, knn);
  k_build_edges<<<dim3(NKE/256, NB), dim3(256), 0, stream>>>(knn, valid, srcb, dstb, emb);

  k_zero<<<dim3((NB*NN*HID)/256), dim3(256), 0, stream>>>(hA, NB*NN*HID);
  k_edgeconv<<<dim3(EDG/64, NB), dim3(128), 0, stream>>>(
      hC, srcb, dstb, emb,
      c2_w1, c2_b1, c2_g1, c2_be1, c2_w2, c2_b2, c2_g2, c2_be2, hA);

  k_gmax<<<dim3(NB), dim3(HID), 0, stream>>>(hA, valid, gfeat);
  k_outmlp<<<dim3(1), dim3(NB), 0, stream>>>(
      gfeat, o_w1, o_b1, o_g1, o_be1, o_w2, o_b2, o_w3, o_b3, outp);
}
